// DCRNN_77257871721177
// MI455X (gfx1250) — matrix operand order repaired and checked by reading, not compiled, not run on hardware
//
#include <hip/hip_runtime.h>
#include <math.h>

// ---------------------------------------------------------------------------
// Types for CDNA5 WMMA (wave32)
// ---------------------------------------------------------------------------
typedef __attribute__((ext_vector_type(16))) __bf16 v16bf;
typedef __attribute__((ext_vector_type(8)))  float  v8f;
typedef __attribute__((ext_vector_type(4)))  unsigned int u32x4;
typedef __attribute__((ext_vector_type(8)))  unsigned int u32x8;

#define N_NODE 512
#define BATCH  64
#define HIDDEN 64
#define BNROWS (BATCH * N_NODE)            // 32768
#define MAXF   128
#define SLAB   ((size_t)N_NODE * BATCH * MAXF)  // elems per diffusion-matrix slab
#define T_IN_  12
#define T_OUT_ 12

// ---------------------------------------------------------------------------
// float -> bf16 (round to nearest even)
// ---------------------------------------------------------------------------
__device__ __forceinline__ unsigned short f2bf(float f) {
    union { float f; unsigned int u; } x; x.f = f;
    unsigned int r = x.u + 0x7FFFu + ((x.u >> 16) & 1u);
    return (unsigned short)(r >> 16);
}

// LDS byte offset of a __shared__ object (generic -> AS(3) -> 32-bit offset)
__device__ __forceinline__ unsigned lds_off_of(const void* p) {
    return (unsigned)(size_t)(__attribute__((address_space(3))) const void*)p;
}

__global__ void k_cvt(const float* __restrict__ src, unsigned short* __restrict__ dst, int n) {
    int i = blockIdx.x * 256 + threadIdx.x;
    if (i < n) dst[i] = f2bf(src[i]);
}

// weight convert with K padding: src (K x O) -> dst (Kpad x O), zeros beyond K
__global__ void k_cvt_pad(const float* __restrict__ src, unsigned short* __restrict__ dst,
                          int K, int O, int Kpad) {
    int i = blockIdx.x * 256 + threadIdx.x;
    int tot = Kpad * O;
    if (i >= tot) return;
    int k = i / O;
    dst[i] = (k < K) ? f2bf(src[i]) : (unsigned short)0;
}

// build x0 = concat(x_in, state) laid out (N, B, F); write f32 + bf16 copies
__global__ void k_concat(const float* __restrict__ xin, int Ci,
                         const float* __restrict__ st,
                         float* __restrict__ x0f, unsigned short* __restrict__ m0, int F) {
    int i = blockIdx.x * 256 + threadIdx.x;
    int tot = N_NODE * BATCH * F;
    if (i >= tot) return;
    int n = i / (BATCH * F);
    int r = i - n * BATCH * F;
    int b = r / F;
    int f = r - b * F;
    float v = (f < Ci) ? xin[((size_t)b * N_NODE + n) * Ci + f]
                       : st[((size_t)b * N_NODE + n) * HIDDEN + (f - Ci)];
    x0f[i] = v;
    m0[i]  = f2bf(v);
}

// pack 5 diffusion matrices (N,B,F) into Xcat (BN x Kpad), k = f*5 + m, zero-pad
__global__ void k_pack(const unsigned short* __restrict__ mats,
                       unsigned short* __restrict__ xcat, int F, int Kpad) {
    int i = blockIdx.x * 256 + threadIdx.x;
    int tot = BNROWS * Kpad;
    if (i >= tot) return;
    int row = i / Kpad;
    int k   = i - row * Kpad;
    unsigned short v = 0;
    int K5 = F * 5;
    if (k < K5) {
        int f = k / 5, m = k - f * 5;
        int b = row >> 9;       // / 512
        int n = row & 511;
        v = mats[(size_t)m * SLAB + ((size_t)n * BATCH + b) * F + f];
    }
    xcat[i] = v;
}

__global__ void k_rstate(const float* __restrict__ g, const float* __restrict__ s,
                         float* __restrict__ rs) {
    int i = blockIdx.x * 256 + threadIdx.x;
    if (i >= BNROWS * HIDDEN) return;
    int row = i >> 6, d = i & 63;
    rs[i] = g[(size_t)row * 128 + d] * s[i];
}

__global__ void k_gru(const float* __restrict__ g, const float* __restrict__ c,
                      float* __restrict__ s) {
    int i = blockIdx.x * 256 + threadIdx.x;
    if (i >= BNROWS * HIDDEN) return;
    int row = i >> 6, d = i & 63;
    float u = g[(size_t)row * 128 + 64 + d];
    s[i] = u * s[i] + (1.0f - u) * c[i];
}

// out = s1 @ p_w + p_b ; also feed back as next decoder input
__global__ void k_projout(const float* __restrict__ s1, const float* __restrict__ pw,
                          const float* __restrict__ pb,
                          float* __restrict__ outT, float* __restrict__ din) {
    int row = blockIdx.x * 256 + threadIdx.x;
    if (row >= BNROWS) return;
    float acc = pb[0];
    const float* sp = s1 + (size_t)row * HIDDEN;
#pragma unroll
    for (int d = 0; d < HIDDEN; ++d) acc += sp[d] * pw[d];
    outT[row] = acc;
    din[row]  = acc;
}

// ---------------------------------------------------------------------------
// TDM: DMA a 32(K) x 64(col) bf16 tile from row-major global (leading dim =
// ldb elements) into LDS *transposed* to [col][k] layout, i.e. the TDM walks
// a 3-D tile with tile_dim0=1 elem, tile_dim1=32 (K, stride ldb),
// tile_dim2=64 (cols, stride 1); LDS fills linearly -> Bs[col][k].
// Tensor dims are set huge so no OOB clamping occurs (tiles are in-range).
// ---------------------------------------------------------------------------
__device__ __forceinline__ void tdm_stage_tr(unsigned ldsOff,
                                             const unsigned short* g,
                                             unsigned ldbElems) {
    unsigned long long ga = (unsigned long long)(size_t)g;
    const unsigned BIG = 0x40000000u;                 // 2^30 elements
    u32x4 d0;
    d0[0] = 1u;                                       // count=1, user descriptor
    d0[1] = ldsOff;                                   // lds_addr
    d0[2] = (unsigned)ga;                             // global_addr[31:0]
    d0[3] = (unsigned)(ga >> 32) | (2u << 30);        // global_addr[56:32] | type=2
    u32x8 d1;
    d1[0] = (1u << 16);                               // data_size=1 (2 bytes)
    d1[1] = (BIG & 0xffffu) << 16;                    // tensor_dim0[15:0]
    d1[2] = ((BIG >> 16) & 0xffffu) | ((BIG & 0xffffu) << 16);  // td0 hi | td1 lo
    d1[3] = ((BIG >> 16) & 0xffffu) | (1u << 16);     // td1 hi | tile_dim0=1
    d1[4] = 32u | (64u << 16);                        // tile_dim1=32 | tile_dim2=64
    d1[5] = ldbElems;                                 // tensor_dim0_stride[31:0]
    d1[6] = (1u << 16);                               // stride0 hi | stride1 lo (=1)
    d1[7] = 0u;                                       // stride1 hi
    u32x4 d2;
    d2[0] = BIG; d2[1] = 0u; d2[2] = 0u; d2[3] = 0u;  // tensor_dim2 big; rest unused
    u32x4 d3;
    d3[0] = 0u; d3[1] = 0u; d3[2] = 0u; d3[3] = 0u;
    asm volatile("tensor_load_to_lds %0, %1, %2, %3"
                 :: "s"(d0), "s"(d1), "s"(d2), "s"(d3)
                 : "memory");
}

// ---------------------------------------------------------------------------
// Tiled bf16 WMMA GEMM: C(M x Ncols) = A(M x K) * B(K x Ncols)
//   block = 256 threads (8 waves), block tile 128 x 64, K step 32
//   wave w -> rows [16w, 16w+16), four 16x16 col tiles (4 accumulators)
//   A fragments loaded straight from global (2 x b128 per lane, ISA A layout)
//   B tiles DMA'd transposed into double-buffered LDS by the Tensor Data
//   Mover (overlapped with compute, s_wait_tensorcnt), fragments read as
//   2 x ds_load_b128
// modes: 0 plain (bf16 out), 1 cheby 2*y-prev (bf16 out),
//        2 bias+sigmoid (f32 out), 3 bias+tanh (f32 out)
// ---------------------------------------------------------------------------
__global__ void __launch_bounds__(256)
k_gemm(const unsigned short* __restrict__ Am, int lda,
       const unsigned short* __restrict__ Bm, int ldb,
       int Kdim, int Ncols,
       float* __restrict__ outF, unsigned short* __restrict__ outBf,
       const float* __restrict__ prevF, const float* __restrict__ bias,
       int mode) {
    __shared__ __align__(16) unsigned short Bs[2][64][32];   // [buf][col][k]

    const int tid  = threadIdx.x;
    const int w    = tid >> 5;
    const int lane = tid & 31;
    const int l15  = lane & 15;
    const int hi   = (lane >> 4) & 1;
    const int rowbase = blockIdx.y * 128 + w * 16;
    const int colbase = blockIdx.x * 64;

    v8f acc[4];
#pragma unroll
    for (int t = 0; t < 4; ++t) acc[t] = (v8f){0.f,0.f,0.f,0.f,0.f,0.f,0.f,0.f};

    union Frag { uint4 q[2]; v16bf v; };

    const unsigned short* aptr = Am + (size_t)(rowbase + l15) * lda;
    const unsigned ldsB0 = lds_off_of(&Bs[0][0][0]);
    const unsigned ldsB1 = lds_off_of(&Bs[1][0][0]);

    // prologue: DMA first B tile (wave 0 only; TDM is a per-wave DMA op)
    if (tid < 32) {
        tdm_stage_tr(ldsB0, Bm + colbase, (unsigned)ldb);
    }

    for (int k0 = 0; k0 < Kdim; k0 += 32) {
        const int cur = (k0 >> 5) & 1;
        if (tid < 32) {
            if (k0 + 32 < Kdim) {
                // DMA next tile into the other buffer, then wait for current
                tdm_stage_tr(cur ? ldsB0 : ldsB1,
                             Bm + (size_t)(k0 + 32) * ldb + colbase, (unsigned)ldb);
                __builtin_amdgcn_s_wait_tensorcnt(1);
            } else {
                __builtin_amdgcn_s_wait_tensorcnt(0);
            }
        }
        __syncthreads();   // current buffer visible to all waves

        Frag fa;
        {
            int klo = k0 + (hi ? 8 : 0);
            int khi = k0 + 16 + (hi ? 8 : 0);
            fa.q[0] = *(const uint4*)(aptr + klo);
            fa.q[1] = *(const uint4*)(aptr + khi);
        }
        const int kst = hi ? 8 : 0;
#pragma unroll
        for (int tx = 0; tx < 4; ++tx) {
            Frag fb;
            const uint4* p = (const uint4*)&Bs[cur][tx * 16 + l15][kst];
            fb.q[0] = p[0];
            fb.q[1] = p[2];
            acc[tx] = __builtin_amdgcn_wmma_f32_16x16x32_bf16(
                false, fa.v, false, fb.v, (short)0, acc[tx], false, false);
        }
        __syncthreads();   // reads done before the buffer is overwritten
    }

    // epilogue
#pragma unroll
    for (int r = 0; r < 8; ++r) {
        int gr = rowbase + r + hi * 8;
#pragma unroll
        for (int tx = 0; tx < 4; ++tx) {
            float v  = acc[tx][r];
            int   gc = colbase + tx * 16 + l15;
            size_t idx = (size_t)gr * Ncols + gc;
            if (mode == 1)      v = 2.0f * v - prevF[idx];
            else if (mode == 2) v = 1.0f / (1.0f + expf(-(v + bias[gc])));
            else if (mode == 3) v = tanhf(v + bias[gc]);
            if (outF)  outF[idx]  = v;
            if (outBf) outBf[idx] = f2bf(v);
        }
    }
}

// ---------------------------------------------------------------------------
// Host orchestration
// ---------------------------------------------------------------------------
static inline void launch_gemm(hipStream_t s, const unsigned short* A, int lda,
                               const unsigned short* B, int ldb, int K, int Ncols, int Mrows,
                               float* outF, unsigned short* outBf, const float* prev,
                               const float* bias, int mode) {
    dim3 g(Ncols / 64, Mrows / 128);
    k_gemm<<<g, 256, 0, s>>>(A, lda, B, ldb, K, Ncols, outF, outBf, prev, bias, mode);
}

extern "C" void kernel_launch(void* const* d_in, const int* in_sizes, int n_in,
                              void* d_out, int out_size, void* d_ws, size_t ws_size,
                              hipStream_t stream) {
    (void)in_sizes; (void)n_in; (void)out_size; (void)ws_size;

    const float* x     = (const float*)d_in[0];
    const float* Amat  = (const float*)d_in[1];
    const float* e0_gw = (const float*)d_in[2];  const float* e0_gb = (const float*)d_in[3];
    const float* e0_cw = (const float*)d_in[4];  const float* e0_cb = (const float*)d_in[5];
    const float* e1_gw = (const float*)d_in[6];  const float* e1_gb = (const float*)d_in[7];
    const float* e1_cw = (const float*)d_in[8];  const float* e1_cb = (const float*)d_in[9];
    const float* d0_gw = (const float*)d_in[10]; const float* d0_gb = (const float*)d_in[11];
    const float* d0_cw = (const float*)d_in[12]; const float* d0_cb = (const float*)d_in[13];
    const float* d1_gw = (const float*)d_in[14]; const float* d1_gb = (const float*)d_in[15];
    const float* d1_cw = (const float*)d_in[16]; const float* d1_cb = (const float*)d_in[17];
    const float* p_w   = (const float*)d_in[18]; const float* p_b   = (const float*)d_in[19];

    // ---- carve workspace -------------------------------------------------
    char* wp = (char*)d_ws;
    auto carve = [&](size_t bytes) -> void* {
        void* p = (void*)wp;
        wp += (bytes + 255) & ~(size_t)255;
        return p;
    };
    unsigned short* Abf      = (unsigned short*)carve((size_t)2 * 512 * 512 * 2);
    unsigned short* e0_gw_bf = (unsigned short*)carve((size_t)352 * 128 * 2);
    unsigned short* e0_cw_bf = (unsigned short*)carve((size_t)352 * 64 * 2);
    unsigned short* e1_gw_bf = (unsigned short*)carve((size_t)640 * 128 * 2);
    unsigned short* e1_cw_bf = (unsigned short*)carve((size_t)640 * 64 * 2);
    unsigned short* d0_gw_bf = (unsigned short*)carve((size_t)352 * 128 * 2);
    unsigned short* d0_cw_bf = (unsigned short*)carve((size_t)352 * 64 * 2);
    unsigned short* d1_gw_bf = (unsigned short*)carve((size_t)640 * 128 * 2);
    unsigned short* d1_cw_bf = (unsigned short*)carve((size_t)640 * 64 * 2);
    float*          x0f      = (float*)carve(SLAB * 4);              // 16 MB
    unsigned short* mats     = (unsigned short*)carve(5 * SLAB * 2); // 40 MB
    unsigned short* xcat     = (unsigned short*)carve((size_t)BNROWS * 640 * 2); // 40 MB
    float*          gbuf     = (float*)carve((size_t)BNROWS * 128 * 4);
    float*          cbuf     = (float*)carve((size_t)BNROWS * 64 * 4);
    float*          rsbuf    = (float*)carve((size_t)BNROWS * 64 * 4);
    float*          s0       = (float*)carve((size_t)BNROWS * 64 * 4); // enc0 / dec0 state
    float*          s1       = (float*)carve((size_t)BNROWS * 64 * 4); // enc1 / dec1 state
    float*          din      = (float*)carve((size_t)BNROWS * 4);

    // ---- one-time conversions -------------------------------------------
    {
        int n = 2 * 512 * 512;
        k_cvt<<<(n + 255) / 256, 256, 0, stream>>>(Amat, Abf, n);
        auto cvtw = [&](const float* src, unsigned short* dst, int K, int O, int Kpad) {
            int tot = Kpad * O;
            k_cvt_pad<<<(tot + 255) / 256, 256, 0, stream>>>(src, dst, K, O, Kpad);
        };
        cvtw(e0_gw, e0_gw_bf, 330, 128, 352);
        cvtw(e0_cw, e0_cw_bf, 330,  64, 352);
        cvtw(e1_gw, e1_gw_bf, 640, 128, 640);
        cvtw(e1_cw, e1_cw_bf, 640,  64, 640);
        cvtw(d0_gw, d0_gw_bf, 325, 128, 352);
        cvtw(d0_cw, d0_cw_bf, 325,  64, 352);
        cvtw(d1_gw, d1_gw_bf, 640, 128, 640);
        cvtw(d1_cw, d1_cw_bf, 640,  64, 640);
    }
    hipMemsetAsync(s0, 0, (size_t)BNROWS * 64 * 4, stream);
    hipMemsetAsync(s1, 0, (size_t)BNROWS * 64 * 4, stream);

    // ---- diffusion conv: state -> activated projection -------------------
    auto dconv = [&](const float* xin, int Ci, const float* state,
                     const unsigned short* w_bf, int Kpad, int O,
                     const float* bias, int actmode, float* outG) {
        int F = Ci + HIDDEN;
        int P = BATCH * F;
        int tot = N_NODE * BATCH * F;
        k_concat<<<(tot + 255) / 256, 256, 0, stream>>>(xin, Ci, state, x0f, mats, F);
        for (int ai = 0; ai < 2; ++ai) {
            const unsigned short* a = Abf + (size_t)ai * 512 * 512;
            unsigned short* m1 = mats + (size_t)(1 + 2 * ai) * SLAB;
            unsigned short* m2 = mats + (size_t)(2 + 2 * ai) * SLAB;
            // x1 = A @ x0
            launch_gemm(stream, a, 512, mats, P, 512, P, 512,
                        nullptr, m1, nullptr, nullptr, 0);
            // x2 = 2*A @ x1 - x0
            launch_gemm(stream, a, 512, m1, P, 512, P, 512,
                        nullptr, m2, x0f, nullptr, 1);
        }
        int ptot = BNROWS * Kpad;
        k_pack<<<(ptot + 255) / 256, 256, 0, stream>>>(mats, xcat, F, Kpad);
        launch_gemm(stream, xcat, Kpad, w_bf, O, Kpad, O, BNROWS,
                    outG, nullptr, nullptr, bias, actmode);
    };

    // ---- DCGRU cell: state updated in place ------------------------------
    auto gru = [&](const float* xin, int Ci, float* state,
                   const unsigned short* gw_bf, const float* gb,
                   const unsigned short* cw_bf, const float* cb, int Kpad) {
        dconv(xin, Ci, state, gw_bf, Kpad, 128, gb, 2, gbuf);          // sigmoid gates
        int n = BNROWS * HIDDEN;
        k_rstate<<<(n + 255) / 256, 256, 0, stream>>>(gbuf, state, rsbuf);
        dconv(xin, Ci, rsbuf, cw_bf, Kpad, 64, cb, 3, cbuf);           // tanh candidate
        k_gru<<<(n + 255) / 256, 256, 0, stream>>>(gbuf, cbuf, state);
    };

    // ---- encoder (layers interleaved per step; no seq storage) -----------
    for (int t = 0; t < T_IN_; ++t) {
        const float* xt = x + (size_t)t * BATCH * N_NODE * 2;
        gru(xt, 2, s0, e0_gw_bf, e0_gb, e0_cw_bf, e0_cb, 352);
        gru(s0, 64, s1, e1_gw_bf, e1_gb, e1_cw_bf, e1_cb, 640);
    }

    // ---- decoder (autoregressive; carry = encoder final states) ----------
    hipMemsetAsync(din, 0, (size_t)BNROWS * 4, stream);
    float* out = (float*)d_out;
    for (int t = 0; t < T_OUT_; ++t) {
        gru(din, 1, s0, d0_gw_bf, d0_gb, d0_cw_bf, d0_cb, 352);
        gru(s0, 64, s1, d1_gw_bf, d1_gb, d1_cw_bf, d1_cb, 640);
        k_projout<<<(BNROWS + 255) / 256, 256, 0, stream>>>(
            s1, p_w, p_b, out + (size_t)t * BNROWS, din);
    }
}
